// SmallThinkerAttention_79121887527467
// MI455X (gfx1250) — matrix operand order repaired and checked by reading, not compiled, not run on hardware
//
#include <hip/hip_runtime.h>
#include <hip/hip_bf16.h>

#define T_TOK   2048
#define HIDDEN  2048
#define NH      32
#define NKV     4
#define HD      64
#define QSZ     (NH * HD)          // 2048
#define KVSZ    (NKV * HD)         // 256
#define QKVW    (QSZ + 2 * KVSZ)   // 2560
#define WINDOW  1024
#define SCALE_Q 0.125f             // HD^-0.5

typedef __attribute__((ext_vector_type(16))) __bf16 v16bf;
typedef __attribute__((ext_vector_type(8)))  float  v8f;
typedef unsigned short u16t;

struct __align__(16) bf16x8 { u16t h[8]; };
union FragBF { v16bf v; bf16x8 c[2]; u16t s[16]; };

__device__ __forceinline__ u16t f2bf(float f) {
  union { float f; unsigned u; } x; x.f = f;
  unsigned u = x.u;
  u += 0x7fffu + ((u >> 16) & 1u);   // round-to-nearest-even
  return (u16t)(u >> 16);
}
__device__ __forceinline__ unsigned pack2bf(float a, float b) {
  return (unsigned)f2bf(a) | ((unsigned)f2bf(b) << 16);
}

// ---------------------------------------------------------------------------
// Tiled bf16-WMMA GEMM: C[M,N] (f32) = A[M,K] * B[K,N]
// A is f32 or bf16 (template), B is f32; both staged in LDS as bf16.
// Block: 256 threads = 8 waves (2 M x 4 N), wave tile 64x32 -> block 128x128.
// Register-double-buffered staging: global loads for tile k+1 issue right
// after the barrier, overlapping the fragment loads + 8 WMMAs of tile k.
// ---------------------------------------------------------------------------
template <bool A_BF16>
__global__ __launch_bounds__(256) void gemm_wmma(
    const void* __restrict__ Ap, const float* __restrict__ B,
    float* __restrict__ C, int M, int N, int K)
{
  constexpr int LDT = 40;   // padded stride (u16): bank-conflict-free b128 frags
  __shared__ __align__(16) u16t As[128 * LDT];
  __shared__ __align__(16) u16t Bs[128 * LDT];

  const int tid  = threadIdx.x;
  const int wave = tid >> 5, lane = tid & 31;
  const int wm = wave >> 2, wn = wave & 3;        // 2 (M) x 4 (N) waves
  const int lm = lane & 15, hi = lane >> 4;
  const int m0 = blockIdx.y * 128, n0 = blockIdx.x * 128;

  // staging assignment: A 128x32 (8 float2/thread), B 32x128 (8 k-pairs/thread)
  const int ac = (tid & 15) * 2;                  // A col pair
  const int ar = tid >> 4;                        // A row, + p*16
  const int bn = tid & 127;                       // B col
  const int bk = (tid >> 7) * 2;                  // B k-pair base, + p*4

  v8f acc[4][2];
#pragma unroll
  for (int i = 0; i < 4; ++i)
#pragma unroll
    for (int j = 0; j < 2; ++j)
      acc[i][j] = (v8f){0.f, 0.f, 0.f, 0.f, 0.f, 0.f, 0.f, 0.f};

  // register staging buffers (next tile)
  float2   aF[8];
  unsigned aB[8];
  float    bF[8][2];

  // prologue: load tile k0 = 0
#pragma unroll
  for (int p = 0; p < 8; ++p) {
    if (A_BF16)
      aB[p] = *(const unsigned*)((const u16t*)Ap + (size_t)(m0 + ar + p * 16) * K + ac);
    else
      aF[p] = *(const float2*)((const float*)Ap + (size_t)(m0 + ar + p * 16) * K + ac);
  }
#pragma unroll
  for (int p = 0; p < 8; ++p) {
    const float* s = B + (size_t)(bk + p * 4) * N + n0 + bn;
    bF[p][0] = s[0];
    bF[p][1] = s[N];
  }

  for (int k0 = 0; k0 < K; k0 += 32) {
    // commit staged registers (tile k0) to LDS as packed bf16
#pragma unroll
    for (int p = 0; p < 8; ++p) {
      unsigned v = A_BF16 ? aB[p] : pack2bf(aF[p].x, aF[p].y);
      *(unsigned*)&As[(ar + p * 16) * LDT + ac] = v;
    }
#pragma unroll
    for (int p = 0; p < 8; ++p)
      *(unsigned*)&Bs[bn * LDT + bk + p * 4] = pack2bf(bF[p][0], bF[p][1]);
    __syncthreads();

    // issue next tile's global loads early (consumed after next barrier)
    if (k0 + 32 < K) {
      const int kn = k0 + 32;
#pragma unroll
      for (int p = 0; p < 8; ++p) {
        if (A_BF16)
          aB[p] = *(const unsigned*)((const u16t*)Ap + (size_t)(m0 + ar + p * 16) * K + kn + ac);
        else
          aF[p] = *(const float2*)((const float*)Ap + (size_t)(m0 + ar + p * 16) * K + kn + ac);
      }
#pragma unroll
      for (int p = 0; p < 8; ++p) {
        const float* s = B + (size_t)(kn + bk + p * 4) * N + n0 + bn;
        bF[p][0] = s[0];
        bF[p][1] = s[N];
      }
      if (k0 + 64 < K)   // pull tile k0+64 toward cache (global_prefetch_b8)
        __builtin_prefetch(B + (size_t)(k0 + 64 + bk) * N + n0 + bn, 0, 1);
    }

    // fragments + 8 WMMAs
    FragBF af[4], bfr[2];
#pragma unroll
    for (int i = 0; i < 4; ++i) {   // A frag: row lm, K chunks hi*8 and 16+hi*8
      const u16t* p = &As[(wm * 64 + i * 16 + lm) * LDT + hi * 8];
      af[i].c[0] = *(const bf16x8*)p;
      af[i].c[1] = *(const bf16x8*)(p + 16);
    }
#pragma unroll
    for (int j = 0; j < 2; ++j) {   // B frag: col lm, K chunks hi*8 and 16+hi*8
      const u16t* p = &Bs[(wn * 32 + j * 16 + lm) * LDT + hi * 8];
      bfr[j].c[0] = *(const bf16x8*)p;
      bfr[j].c[1] = *(const bf16x8*)(p + 16);
    }
#pragma unroll
    for (int i = 0; i < 4; ++i)
#pragma unroll
      for (int j = 0; j < 2; ++j)
        acc[i][j] = __builtin_amdgcn_wmma_f32_16x16x32_bf16(
            false, af[i].v, false, bfr[j].v, (short)0, acc[i][j], false, false);
    __syncthreads();
  }

#pragma unroll
  for (int i = 0; i < 4; ++i)
#pragma unroll
    for (int j = 0; j < 2; ++j)
#pragma unroll
      for (int r = 0; r < 8; ++r)
        C[(size_t)(m0 + wm * 64 + i * 16 + hi * 8 + r) * N +
          n0 + wn * 32 + j * 16 + lm] = acc[i][j][r];
}

// ---------------------------------------------------------------------------
// RoPE + layout kernels
// ---------------------------------------------------------------------------
__global__ __launch_bounds__(256) void rope_q_kernel(
    const float* __restrict__ qkv, const int* __restrict__ pos,
    u16t* __restrict__ qb)
{
  int idx = blockIdx.x * 256 + threadIdx.x;     // T*NH*32
  int d = idx & 31, h = (idx >> 5) & 31, t = idx >> 10;
  const float L2T = 19.9315685693241741f;       // log2(1e6)
  float inv = exp2f(-(float)d * (L2T * (1.0f / 32.0f)));
  float ang = (float)pos[t] * inv;
  float s, c; __sincosf(ang, &s, &c);
  const float* src = qkv + (size_t)t * QKVW + h * HD;
  float x1 = src[d], x2 = src[d + 32];
  u16t* dst = qb + (size_t)t * QSZ + h * HD;
  dst[d]      = f2bf((x1 * c - x2 * s) * SCALE_Q);  // fold softmax scale into Q
  dst[d + 32] = f2bf((x2 * c + x1 * s) * SCALE_Q);
}

__global__ __launch_bounds__(256) void rope_k_kernel(
    const float* __restrict__ qkv, const int* __restrict__ pos,
    u16t* __restrict__ kb)     // kb[kvh][T][HD]
{
  int idx = blockIdx.x * 256 + threadIdx.x;     // T*NKV*32
  int d = idx & 31, h = (idx >> 5) & 3, t = idx >> 7;
  const float L2T = 19.9315685693241741f;
  float inv = exp2f(-(float)d * (L2T * (1.0f / 32.0f)));
  float ang = (float)pos[t] * inv;
  float s, c; __sincosf(ang, &s, &c);
  const float* src = qkv + (size_t)t * QKVW + QSZ + h * HD;
  float x1 = src[d], x2 = src[d + 32];
  u16t* dst = kb + ((size_t)h * T_TOK + t) * HD;
  dst[d]      = f2bf(x1 * c - x2 * s);
  dst[d + 32] = f2bf(x2 * c + x1 * s);
}

__global__ __launch_bounds__(256) void vtrans_kernel(
    const float* __restrict__ qkv, u16t* __restrict__ vt)  // vt[kvh][HD][T]
{
  int idx = blockIdx.x * 256 + threadIdx.x;     // NKV*HD*T, t fastest
  int t = idx & (T_TOK - 1);
  int d = (idx >> 11) & 63;
  int h = idx >> 17;
  vt[(size_t)(h * HD + d) * T_TOK + t] =
      f2bf(qkv[(size_t)t * QKVW + QSZ + KVSZ + h * HD + d]);
}

// ---------------------------------------------------------------------------
// Flash attention: one wave per (head, 16-query tile); 32-key chunks.
// ---------------------------------------------------------------------------
__global__ __launch_bounds__(256) void attn_kernel(
    const u16t* __restrict__ qb,   // [T][QSZ]        (rope'd, pre-scaled)
    const u16t* __restrict__ kb,   // [NKV][T][HD]    (rope'd)
    const u16t* __restrict__ vt,   // [NKV][HD][T]    (transposed)
    u16t* __restrict__ attb)       // [T][QSZ]
{
  __shared__ __align__(16) u16t Ps[8][16 * 40];   // per-wave P tile, padded

  const int wave = threadIdx.x >> 5, lane = threadIdx.x & 31;
  const int lm = lane & 15, hi = lane >> 4;
  const int wg  = blockIdx.x * 8 + wave;
  const int h   = wg >> 7;              // 128 q-tiles per head
  const int qi  = wg & 127;
  const int kvh = h >> 3;               // GQA group = 8
  const int i0  = qi * 16;

  // Q fragments: 16 rows x 64 = two 16x32 A-frags
  FragBF qf[2];
  {
    const u16t* qrow = qb + (size_t)(i0 + lm) * QSZ + h * HD;
#pragma unroll
    for (int kk = 0; kk < 2; ++kk) {
      const u16t* p = qrow + kk * 32 + hi * 8;
      qf[kk].c[0] = *(const bf16x8*)p;
      qf[kk].c[1] = *(const bf16x8*)(p + 16);
    }
  }

  v8f o[4];
#pragma unroll
  for (int n = 0; n < 4; ++n) o[n] = (v8f){0.f,0.f,0.f,0.f,0.f,0.f,0.f,0.f};
  float mr[8], lr[8];
#pragma unroll
  for (int r = 0; r < 8; ++r) { mr[r] = -1e30f; lr[r] = 0.f; }

  int jlo = i0 - (WINDOW - 1); if (jlo < 0) jlo = 0;
  const int jc0 = jlo & ~31;

  for (int jc = jc0; jc <= i0 + 15; jc += 32) {
    float sv[2][8];
#pragma unroll
    for (int jt = 0; jt < 2; ++jt) {
      const int j0 = jc + jt * 16;
      v8f s = (v8f){0.f,0.f,0.f,0.f,0.f,0.f,0.f,0.f};
#pragma unroll
      for (int kk = 0; kk < 2; ++kk) {   // K rows as B-frag (kb is [j][d] = [n][k])
        FragBF kf;
        const u16t* p = kb + ((size_t)kvh * T_TOK + (j0 + lm)) * HD + kk * 32 + hi * 8;
        kf.c[0] = *(const bf16x8*)p;
        kf.c[1] = *(const bf16x8*)(p + 16);
        s = __builtin_amdgcn_wmma_f32_16x16x32_bf16(
            false, qf[kk].v, false, kf.v, (short)0, s, false, false);
      }
      const int j_abs = j0 + lm;
#pragma unroll
      for (int r = 0; r < 8; ++r) {
        const int i_abs = i0 + hi * 8 + r;
        const bool ok = (j_abs <= i_abs) && (i_abs - j_abs < WINDOW);
        sv[jt][r] = ok ? s[r] : -1e30f;
      }
    }
    // online softmax over the 32-key chunk
#pragma unroll
    for (int r = 0; r < 8; ++r) {
      float tm = fmaxf(sv[0][r], sv[1][r]);
      tm = fmaxf(tm, __shfl_xor(tm, 1, 32));
      tm = fmaxf(tm, __shfl_xor(tm, 2, 32));
      tm = fmaxf(tm, __shfl_xor(tm, 4, 32));
      tm = fmaxf(tm, __shfl_xor(tm, 8, 32));
      float mn = fmaxf(fmaxf(mr[r], tm), -1e20f);  // clamp: all-masked chunk -> p=0
      float corr = __expf(mr[r] - mn);
      float p0 = __expf(sv[0][r] - mn);
      float p1 = __expf(sv[1][r] - mn);
      float ts = p0 + p1;
      ts += __shfl_xor(ts, 1, 32);
      ts += __shfl_xor(ts, 2, 32);
      ts += __shfl_xor(ts, 4, 32);
      ts += __shfl_xor(ts, 8, 32);
      lr[r] = lr[r] * corr + ts;
      mr[r] = mn;
#pragma unroll
      for (int n = 0; n < 4; ++n) o[n][r] *= corr;
      Ps[wave][(hi * 8 + r) * 40 + lm]      = f2bf(p0);
      Ps[wave][(hi * 8 + r) * 40 + 16 + lm] = f2bf(p1);
    }
    // P (16x32) as A-frag from LDS (same-wave DS ops are in-order)
    FragBF pf;
    {
      const u16t* pp = &Ps[wave][lm * 40 + hi * 8];
      pf.c[0] = *(const bf16x8*)pp;
      pf.c[1] = *(const bf16x8*)(pp + 16);
    }
#pragma unroll
    for (int n = 0; n < 4; ++n) {        // V^T rows d = B-frag cols, k=j chunks hi*8 and 16+hi*8
      FragBF vf;
      const u16t* vp = vt + (size_t)(kvh * HD + n * 16 + lm) * T_TOK + jc + hi * 8;
      vf.c[0] = *(const bf16x8*)vp;
      vf.c[1] = *(const bf16x8*)(vp + 16);
      o[n] = __builtin_amdgcn_wmma_f32_16x16x32_bf16(
          false, pf.v, false, vf.v, (short)0, o[n], false, false);
    }
  }

#pragma unroll
  for (int r = 0; r < 8; ++r) {
    const float inv = 1.0f / lr[r];
    const int i_abs = i0 + hi * 8 + r;
#pragma unroll
    for (int n = 0; n < 4; ++n)
      attb[(size_t)i_abs * QSZ + h * HD + n * 16 + lm] = f2bf(o[n][r] * inv);
  }
}

// ---------------------------------------------------------------------------
extern "C" void kernel_launch(void* const* d_in, const int* in_sizes, int n_in,
                              void* d_out, int out_size, void* d_ws, size_t ws_size,
                              hipStream_t stream)
{
  (void)in_sizes; (void)n_in; (void)out_size; (void)ws_size;
  const int*   pos    = (const int*)d_in[0];
  const float* hidden = (const float*)d_in[1];
  const float* wqkv   = (const float*)d_in[2];
  const float* wo     = (const float*)d_in[3];
  float* out = (float*)d_out;

  char* ws = (char*)d_ws;
  float* qkv  = (float*)(ws);                 // 2048*2560*4 = 20,971,520 B
  u16t*  qb   = (u16t*)(ws + 20971520);       // 2048*2048*2 =  8,388,608 B
  u16t*  kb   = (u16t*)(ws + 29360128);       // 4*2048*64*2 =  1,048,576 B
  u16t*  vt   = (u16t*)(ws + 30408704);       //                1,048,576 B
  u16t*  attb = (u16t*)(ws + 31457280);       //                8,388,608 B

  // 1) fused QKV projection (f32 in, bf16 WMMA, f32 out)
  gemm_wmma<false><<<dim3(QKVW / 128, T_TOK / 128), 256, 0, stream>>>(
      hidden, wqkv, qkv, T_TOK, QKVW, HIDDEN);
  // 2) RoPE + attention-friendly layouts
  rope_q_kernel<<<(T_TOK * NH * 32) / 256, 256, 0, stream>>>(qkv, pos, qb);
  rope_k_kernel<<<(T_TOK * NKV * 32) / 256, 256, 0, stream>>>(qkv, pos, kb);
  vtrans_kernel<<<(T_TOK * NKV * HD) / 256, 256, 0, stream>>>(qkv, vt);
  // 3) flash attention (sliding window, GQA)
  attn_kernel<<<(NH * (T_TOK / 16)) / 8, 256, 0, stream>>>(qb, kb, vt, attb);
  // 4) output projection (bf16 A path)
  gemm_wmma<true><<<dim3(HIDDEN / 128, T_TOK / 128), 256, 0, stream>>>(
      attb, wo, out, T_TOK, HIDDEN, QSZ);
}
